// BatchGraphVariationalAutoencoder_3504693314191
// MI455X (gfx1250) — matrix operand order repaired and checked by reading, not compiled, not run on hardware
//
#include <hip/hip_runtime.h>
#include <hip/hip_bf16.h>
#include <math.h>

// ---------------------------------------------------------------------------
// CDNA5 (gfx1250) wave32 WMMA GEMM pipeline for the batch graph VAE.
// All GEMMs: v_wmma_f32_16x16x32_bf16, bf16-resident operands, f32 accumulate.
// Block tile 128(M) x 64(N), K-step 32; 8 waves, wave owns 32x32 (4 WMMAs).
// Double-buffered LDS, single barrier per K-step.
// A tile (non-transposed path) staged with GLOBAL_LOAD_ASYNC_TO_LDS_B128
// (ASYNCcnt), removing the VGPR round trip entirely.
// ---------------------------------------------------------------------------

typedef __attribute__((ext_vector_type(16))) __bf16 v16bf;
typedef __attribute__((ext_vector_type(8)))  float  v8f;

#define TM 128
#define TN 64
#define TK 32
#define KPAD 40            // bf16 per LDS row (80B, 16B aligned, 20-dword stride)
#define KPADU 20           // row stride in packed uints

__device__ __forceinline__ float softplusf(float x) {
  return (x > 15.0f) ? x : log1pf(__expf(x));
}

__device__ __forceinline__ void wait_async() {
  asm volatile("s_wait_asynccnt 0x0" ::: "memory");
}

// Async DMA: copy this thread's two 16B chunks of the A tile into LDS.
__device__ __forceinline__ void async_copy_A(const __bf16* __restrict__ A,
                                             int lda, int m0, int kk, int tid,
                                             __bf16 (*Asb)[KPAD])
{
  const int r  = tid >> 1;              // 0..127
  const int kc = (tid & 1) * 16;        // 0 / 16  (bf16 elements)
  const __bf16* g0 = A + (size_t)(m0 + r) * (size_t)lda + (kk + kc);
  const unsigned l0 = (unsigned)(size_t)&Asb[r][kc];   // low 32 bits = LDS offset
  asm volatile("global_load_async_to_lds_b128 %0, %1, off"
               :: "v"(l0), "v"((unsigned long long)(size_t)g0) : "memory");
  asm volatile("global_load_async_to_lds_b128 %0, %1, off"
               :: "v"(l0 + 16u), "v"((unsigned long long)((size_t)g0 + 16)) : "memory");
}

struct StageRegs {
  unsigned ta0[4], ta1[4];     // A tile, TRANSA path
  unsigned tb0[2], tb1[2];     // B tile
};

template<bool TRANSA>
__device__ __forceinline__ void tile_load(const __bf16* __restrict__ A,
                                          const __bf16* __restrict__ B,
                                          int lda, int ldb, int m0, int n0,
                                          int kk, int tid, StageRegs& R)
{
  if (TRANSA) {
    const int m  = (tid & 63) * 2;        // even m 0..126
    const int kq = tid >> 6;              // 0..3
    #pragma unroll
    for (int j = 0; j < 4; ++j) {
      const int kp = kq * 4 + j;          // k-pair 0..15
      R.ta0[j] = *(const unsigned*)(A + (size_t)(kk + 2 * kp)     * (size_t)lda + (m0 + m));
      R.ta1[j] = *(const unsigned*)(A + (size_t)(kk + 2 * kp + 1) * (size_t)lda + (m0 + m));
    }
  }
  {
    const int n  = (tid & 31) * 2;        // even n 0..62
    const int kq = tid >> 5;              // 0..7
    #pragma unroll
    for (int j = 0; j < 2; ++j) {
      const int kp = kq * 2 + j;          // k-pair 0..15
      R.tb0[j] = *(const unsigned*)(B + (size_t)(kk + 2 * kp)     * (size_t)ldb + (n0 + n));
      R.tb1[j] = *(const unsigned*)(B + (size_t)(kk + 2 * kp + 1) * (size_t)ldb + (n0 + n));
    }
  }
}

template<bool TRANSA>
__device__ __forceinline__ void tile_store(__bf16 (*As)[KPAD], __bf16 (*Bs)[KPAD],
                                           int tid, const StageRegs& R)
{
  unsigned* Asu = (unsigned*)&As[0][0];
  unsigned* Bsu = (unsigned*)&Bs[0][0];
  if (TRANSA) {
    const int m  = (tid & 63) * 2;
    const int kq = tid >> 6;
    #pragma unroll
    for (int j = 0; j < 4; ++j) {
      const int kp = kq * 4 + j;
      const unsigned lo = (R.ta0[j] & 0xffffu) | (R.ta1[j] << 16);
      const unsigned hi = (R.ta0[j] >> 16)     | (R.ta1[j] & 0xffff0000u);
      Asu[m * KPADU + kp]       = lo;
      Asu[(m + 1) * KPADU + kp] = hi;
    }
  }
  {
    const int n  = (tid & 31) * 2;
    const int kq = tid >> 5;
    #pragma unroll
    for (int j = 0; j < 2; ++j) {
      const int kp = kq * 2 + j;
      const unsigned lo = (R.tb0[j] & 0xffffu) | (R.tb1[j] << 16);
      const unsigned hi = (R.tb0[j] >> 16)     | (R.tb1[j] & 0xffff0000u);
      Bsu[n * KPADU + kp]       = lo;
      Bsu[(n + 1) * KPADU + kp] = hi;
    }
  }
}

// ACT: 0=none(no bias) 1=+bias 2=+bias,relu 3=+bias,softplus
// OUTF32: true -> C is float*, else C is __bf16*
template<int ACT, bool TRANSA, bool OUTF32>
__global__ __launch_bounds__(256)
void gemm_wmma(const __bf16* __restrict__ A, const __bf16* __restrict__ B,
               const float* __restrict__ bias, void* __restrict__ Cv,
               int K, int lda, int ldb, int ldc,
               long long sA, long long sB, long long sC)
{
  __shared__ __align__(16) __bf16 As[2][TM][KPAD];
  __shared__ __align__(16) __bf16 Bs[2][TN][KPAD];

  const int bz = blockIdx.z;
  A += (size_t)bz * (size_t)sA;
  B += (size_t)bz * (size_t)sB;

  const int m0 = blockIdx.y * TM;
  const int n0 = blockIdx.x * TN;

  const int tid  = threadIdx.x;
  const int w    = tid >> 5;
  const int lane = tid & 31;
  const int half = lane >> 4;
  const int ln   = lane & 15;

  const int wm = (w >> 1) * 32;     // 0,32,64,96
  const int wn = (w & 1) * 32;      // 0,32

  v8f c00 = {}, c01 = {}, c10 = {}, c11 = {};

  StageRegs R;
  if (!TRANSA) async_copy_A(A, lda, m0, 0, tid, As[0]);
  tile_load<TRANSA>(A, B, lda, ldb, m0, n0, 0, tid, R);
  tile_store<TRANSA>(As[0], Bs[0], tid, R);
  if (!TRANSA) wait_async();
  __syncthreads();

  int buf = 0;
  for (int kk = 0; kk < K; kk += TK) {
    const bool has_next = (kk + TK) < K;
    if (has_next) {
      if (!TRANSA) async_copy_A(A, lda, m0, kk + TK, tid, As[buf ^ 1]);
      tile_load<TRANSA>(A, B, lda, ldb, m0, n0, kk + TK, tid, R);
    }

    // ---- fragments per ISA 7.12.2 layouts ----
    v16bf a0, a1, b0, b1;
    #pragma unroll
    for (int v = 0; v < 8; ++v) {
      const int k0 = 2 * v + 8 * half + ((v >= 4) ? 8 : 0);
      a0[2 * v]     = As[buf][wm + ln][k0];
      a0[2 * v + 1] = As[buf][wm + ln][k0 + 1];
      a1[2 * v]     = As[buf][wm + 16 + ln][k0];
      a1[2 * v + 1] = As[buf][wm + 16 + ln][k0 + 1];
    }
    #pragma unroll
    for (int e = 0; e < 16; ++e) {
      b0[e] = Bs[buf][wn + ln][(e & 7) + 8 * half + ((e >> 3) * 16)];
      b1[e] = Bs[buf][wn + 16 + ln][(e & 7) + 8 * half + ((e >> 3) * 16)];
    }

    // staged registers die here (before the accumulator/fragment peak)
    if (has_next) tile_store<TRANSA>(As[buf ^ 1], Bs[buf ^ 1], tid, R);

    c00 = __builtin_amdgcn_wmma_f32_16x16x32_bf16(false, a0, false, b0,
                                                  (short)0, c00, false, false);
    c01 = __builtin_amdgcn_wmma_f32_16x16x32_bf16(false, a0, false, b1,
                                                  (short)0, c01, false, false);
    c10 = __builtin_amdgcn_wmma_f32_16x16x32_bf16(false, a1, false, b0,
                                                  (short)0, c10, false, false);
    c11 = __builtin_amdgcn_wmma_f32_16x16x32_bf16(false, a1, false, b1,
                                                  (short)0, c11, false, false);

    if (!TRANSA) wait_async();
    __syncthreads();
    buf ^= 1;
  }

  // ---- epilogue ----
  #pragma unroll
  for (int i = 0; i < 8; ++i) {
    const int row0 = m0 + wm + 8 * half + i;
    const int row1 = row0 + 16;
    const int col0 = n0 + wn + ln;
    const int col1 = col0 + 16;
    float v00 = c00[i], v01 = c01[i], v10 = c10[i], v11 = c11[i];
    if (ACT >= 1) {
      const float bz0 = bias[col0], bz1 = bias[col1];
      v00 += bz0; v01 += bz1; v10 += bz0; v11 += bz1;
    }
    if (ACT == 2) {
      v00 = fmaxf(v00, 0.0f); v01 = fmaxf(v01, 0.0f);
      v10 = fmaxf(v10, 0.0f); v11 = fmaxf(v11, 0.0f);
    }
    if (ACT == 3) {
      v00 = softplusf(v00); v01 = softplusf(v01);
      v10 = softplusf(v10); v11 = softplusf(v11);
    }
    if (OUTF32) {
      float* C = (float*)Cv + (size_t)bz * (size_t)sC;
      C[(size_t)row0 * (size_t)ldc + col0] = v00;
      C[(size_t)row0 * (size_t)ldc + col1] = v01;
      C[(size_t)row1 * (size_t)ldc + col0] = v10;
      C[(size_t)row1 * (size_t)ldc + col1] = v11;
    } else {
      __bf16* C = (__bf16*)Cv + (size_t)bz * (size_t)sC;
      C[(size_t)row0 * (size_t)ldc + col0] = (__bf16)v00;
      C[(size_t)row0 * (size_t)ldc + col1] = (__bf16)v01;
      C[(size_t)row1 * (size_t)ldc + col0] = (__bf16)v10;
      C[(size_t)row1 * (size_t)ldc + col1] = (__bf16)v11;
    }
  }
}

// ---------------------------------------------------------------------------
// f32 -> bf16 converter (paired stores)
// ---------------------------------------------------------------------------
__global__ __launch_bounds__(256)
void cvt_f32_bf16(const float* __restrict__ in, __bf16* __restrict__ o, int n2)
{
  const int i = blockIdx.x * 256 + threadIdx.x;
  if (i >= n2) return;
  union { __bf16 h[2]; unsigned u; } p;
  p.h[0] = (__bf16)in[2 * i];
  p.h[1] = (__bf16)in[2 * i + 1];
  ((unsigned*)o)[i] = p.u;
}

// ---------------------------------------------------------------------------
// Row softmax on bf16 buffer (f32 math). One 8-wave block per row.
// ---------------------------------------------------------------------------
__global__ __launch_bounds__(256)
void softmax_rows_bf16(__bf16* __restrict__ p, int K)
{
  __bf16* pr = p + (size_t)blockIdx.x * (size_t)K;
  const int tid = threadIdx.x;
  __shared__ float red[8];

  float m = -3.0e38f;
  for (int i = tid; i < K; i += 256) m = fmaxf(m, (float)pr[i]);
  #pragma unroll
  for (int off = 16; off; off >>= 1) m = fmaxf(m, __shfl_xor(m, off, 32));
  if ((tid & 31) == 0) red[tid >> 5] = m;
  __syncthreads();
  #pragma unroll
  for (int i = 0; i < 8; ++i) m = fmaxf(m, red[i]);
  __syncthreads();

  float s = 0.0f;
  for (int i = tid; i < K; i += 256) {
    const float e = __expf((float)pr[i] - m);
    pr[i] = (__bf16)e;
    s += e;
  }
  #pragma unroll
  for (int off = 16; off; off >>= 1) s += __shfl_xor(s, off, 32);
  if ((tid & 31) == 0) red[tid >> 5] = s;
  __syncthreads();
  s = 0.0f;
  #pragma unroll
  for (int i = 0; i < 8; ++i) s += red[i];
  const float inv = 1.0f / s;
  for (int i = tid; i < K; i += 256) pr[i] = (__bf16)((float)pr[i] * inv);
}

// ---------------------------------------------------------------------------
// Reparameterize: g (f32 [B,512,128]) -> mean/log_var (f32) + z (bf16)
// ---------------------------------------------------------------------------
__global__ __launch_bounds__(256)
void reparam(const float* __restrict__ g, const float* __restrict__ eps,
             float* __restrict__ mean_out, float* __restrict__ logv_out,
             __bf16* __restrict__ z, int total)
{
  const int i = blockIdx.x * 256 + threadIdx.x;
  if (i >= total) return;
  const int c  = i & 63;        // LAT = 64
  const int bn = i >> 6;        // b*512 + n
  const float mu = g[(size_t)bn * 128 + c];
  const float lv = g[(size_t)bn * 128 + 64 + c];
  mean_out[i] = mu;
  logv_out[i] = lv;
  z[i] = (__bf16)(mu + __expf(0.5f * lv) * eps[i]);
}

// ---------------------------------------------------------------------------
// Host side
// ---------------------------------------------------------------------------
template<int ACT, bool TA, bool OF32>
static inline void G(hipStream_t st, const __bf16* A, const __bf16* B,
                     const float* bias, void* C,
                     int M, int N, int K, int lda, int ldb, int ldc,
                     long long sA, long long sB, long long sC)
{
  dim3 grid(N / TN, M / TM, 8);
  gemm_wmma<ACT, TA, OF32><<<grid, dim3(256), 0, st>>>(A, B, bias, C, K,
                                                       lda, ldb, ldc, sA, sB, sC);
}

extern "C" void kernel_launch(void* const* d_in, const int* in_sizes, int n_in,
                              void* d_out, int out_size, void* d_ws, size_t ws_size,
                              hipStream_t stream)
{
  (void)in_sizes; (void)n_in; (void)out_size; (void)ws_size;

  const float* x      = (const float*)d_in[0];   // [8,2048,64]
  const float* eps    = (const float*)d_in[1];   // [8,512,64]
  const float* adj    = (const float*)d_in[2];   // [2048,2048] shared
  const float* We1    = (const float*)d_in[3];
  const float* be1    = (const float*)d_in[4];
  const float* Kemb1  = (const float*)d_in[5];
  const float* Kpool1 = (const float*)d_in[6];
  const float* We2    = (const float*)d_in[7];
  const float* be2    = (const float*)d_in[8];
  const float* Kemb2  = (const float*)d_in[9];
  const float* Kpool2 = (const float*)d_in[10];
  const float* We3    = (const float*)d_in[11];
  const float* be3    = (const float*)d_in[12];
  const float* Wd0    = (const float*)d_in[13];
  const float* bd0    = (const float*)d_in[14];
  const float* Wd1    = (const float*)d_in[15];
  const float* bd1    = (const float*)d_in[16];
  const float* Wd2    = (const float*)d_in[17];
  const float* bd2    = (const float*)d_in[18];
  const float* Wdf    = (const float*)d_in[19];
  const float* bdf    = (const float*)d_in[20];

  float*  out = (float*)d_out;           // [8,2048,64] + mean + log_var
  __bf16* wb  = (__bf16*)d_ws;

  // dims
  const int Nn = 2048, Hh = 256, K1 = 1024, K2 = 512;
  const long long sXH  = (long long)Nn * Hh;
  const long long sS1  = (long long)Nn * K1;
  const long long sA1  = (long long)K1 * K1;
  const long long sK1H = (long long)K1 * Hh;
  const long long sS2  = (long long)K1 * K2;
  const long long sA2  = (long long)K2 * K2;
  const long long sK2H = (long long)K2 * Hh;
  const long long sG   = (long long)K2 * 128;
  const long long sZL  = (long long)K2 * 64;
  const long long sOUT = (long long)Nn * 64;

  // ---- workspace layout (bf16 elements), manual reuse ----
  __bf16* T1  = wb;                     // 4,194,304  temps (X@W)
  __bf16* H   = wb + 4194304;           // 4,194,304  h1 / h2 / d3
  __bf16* Z1  = wb + 8388608;           // 4,194,304  z1 / z2 / u2
  __bf16* TB  = wb + 12582912;          // 16,777,216 t2/AS1/t5/AS2/u1
  __bf16* D1  = TB + 4194304;           //            d1 (inside TB)
  __bf16* S1  = wb + 29360128;          // 16,777,216 p1 -> s1
  __bf16* A1  = wb + 46137344;          // 8,388,608
  __bf16* HP  = wb + 54525952;          // 2,097,152  hp1 / d2
  __bf16* S2  = wb + 56623104;          // 4,194,304  p2 -> s2
  __bf16* A2  = wb + 60817408;          // 2,097,152
  __bf16* HP2 = wb + 62914560;          // 1,048,576
  __bf16* ZL  = wb + 63963136;          //   262,144
  __bf16* ADJ = wb + 64225280;          // 4,194,304  adj bf16
  __bf16* XH  = wb + 68419584;          // 1,048,576  x bf16
  __bf16* cw  = wb + 69468160;          // converted weights:
  __bf16* We1h    = cw;                 //    16,384
  __bf16* Kemb1h  = cw + 16384;         //    65,536
  __bf16* Kpool1h = cw + 81920;         //   262,144
  __bf16* We2h    = cw + 344064;        //    65,536
  __bf16* Kemb2h  = cw + 409600;        //    65,536
  __bf16* Kpool2h = cw + 475136;        //   131,072
  __bf16* We3h    = cw + 606208;        //    32,768
  __bf16* Wd0h    = cw + 638976;        //    16,384
  __bf16* Wd1h    = cw + 655360;        //    65,536
  __bf16* Wd2h    = cw + 720896;        //    65,536
  __bf16* Wdfh    = cw + 786432;        //    16,384
  float* Gb = (float*)(wb + 70270976);  //   524,288 f32 (g = mean|log_var)

  auto CVT = [&](const float* src, __bf16* dst, int n) {
    cvt_f32_bf16<<<dim3((n / 2 + 255) / 256), dim3(256), 0, stream>>>(src, dst, n / 2);
  };

  // ---- one-time (per call) f32 -> bf16 conversion of all GEMM operands ----
  CVT(adj,    ADJ,     Nn * Nn);
  CVT(x,      XH,      8 * Nn * 64);
  CVT(We1,    We1h,    64 * Hh);
  CVT(Kemb1,  Kemb1h,  Hh * Hh);
  CVT(Kpool1, Kpool1h, Hh * K1);
  CVT(We2,    We2h,    Hh * Hh);
  CVT(Kemb2,  Kemb2h,  Hh * Hh);
  CVT(Kpool2, Kpool2h, Hh * K2);
  CVT(We3,    We3h,    Hh * 128);
  CVT(Wd0,    Wd0h,    64 * Hh);
  CVT(Wd1,    Wd1h,    Hh * Hh);
  CVT(Wd2,    Wd2h,    Hh * Hh);
  CVT(Wdf,    Wdfh,    Hh * 64);

  // ---------------- encoder ----------------
  // 1. t0 = x @ We1 ; h1 = relu(A0 @ t0 + be1)
  G<0,false,false>(stream, XH,  We1h, nullptr, T1, Nn, Hh, 64, 64, Hh, Hh, (long long)Nn*64, 0, sXH);
  G<2,false,false>(stream, ADJ, T1,   be1,     H,  Nn, Hh, Nn, Nn, Hh, Hh, 0, sXH, sXH);
  // 2. DiffPool1: z1 = A0 @ (h1 @ Kemb1)
  G<0,false,false>(stream, H,   Kemb1h, nullptr, T1, Nn, Hh, Hh, Hh, Hh, Hh, sXH, 0, sXH);
  G<0,false,false>(stream, ADJ, T1,     nullptr, Z1, Nn, Hh, Nn, Nn, Hh, Hh, 0, sXH, sXH);
  //    s1 = softmax(A0 @ (h1 @ Kpool1))
  G<0,false,false>(stream, H,   Kpool1h, nullptr, TB, Nn, K1, Hh, Hh, K1, K1, sXH, 0, sS1);
  G<0,false,false>(stream, ADJ, TB,      nullptr, S1, Nn, K1, Nn, Nn, K1, K1, 0, sS1, sS1);
  softmax_rows_bf16<<<dim3(8 * Nn), dim3(256), 0, stream>>>(S1, K1);
  //    hp1 = s1^T @ z1 ; AS1 = A0 @ s1 ; A1 = s1^T @ AS1
  G<0,true ,false>(stream, S1,  Z1, nullptr, HP, K1, Hh, Nn, K1, Hh, Hh, sS1, sXH, sK1H);
  G<0,false,false>(stream, ADJ, S1, nullptr, TB, Nn, K1, Nn, Nn, K1, K1, 0, sS1, sS1);
  G<0,true ,false>(stream, S1,  TB, nullptr, A1, K1, K1, Nn, K1, K1, K1, sS1, sS1, sA1);
  // 3. h2 = relu(A1 @ (hp1 @ We2) + be2)
  G<0,false,false>(stream, HP, We2h, nullptr, T1, K1, Hh, Hh, Hh, Hh, Hh, sK1H, 0, sK1H);
  G<2,false,false>(stream, A1, T1,   be2,     H,  K1, Hh, K1, K1, Hh, Hh, sA1, sK1H, sK1H);
  // 4. DiffPool2: z2 = A1 @ (h2 @ Kemb2)
  G<0,false,false>(stream, H,  Kemb2h, nullptr, T1, K1, Hh, Hh, Hh, Hh, Hh, sK1H, 0, sK1H);
  G<0,false,false>(stream, A1, T1,     nullptr, Z1, K1, Hh, K1, K1, Hh, Hh, sA1, sK1H, sK1H);
  //    s2 = softmax(A1 @ (h2 @ Kpool2))
  G<0,false,false>(stream, H,  Kpool2h, nullptr, TB, K1, K2, Hh, Hh, K2, K2, sK1H, 0, sS2);
  G<0,false,false>(stream, A1, TB,      nullptr, S2, K1, K2, K1, K1, K2, K2, sA1, sS2, sS2);
  softmax_rows_bf16<<<dim3(8 * K1), dim3(256), 0, stream>>>(S2, K2);
  //    hp2 = s2^T @ z2 ; AS2 = A1 @ s2 ; A2 = s2^T @ AS2
  G<0,true ,false>(stream, S2, Z1, nullptr, HP2, K2, Hh, K1, K2, Hh, Hh, sS2, sK1H, sK2H);
  G<0,false,false>(stream, A1, S2, nullptr, TB,  K1, K2, K1, K1, K2, K2, sA1, sS2, sS2);
  G<0,true ,false>(stream, S2, TB, nullptr, A2,  K2, K2, K1, K2, K2, K2, sS2, sS2, sA2);
  // 5. g = A2 @ (hp2 @ We3) + be3  (bias only, f32 out)
  G<0,false,false>(stream, HP2, We3h, nullptr, T1, K2, 128, Hh, Hh, 128, 128, sK2H, 0, sG);
  G<1,false,true >(stream, A2,  T1,   be3,     Gb, K2, 128, K2, K2, 128, 128, sA2, sG, sG);
  // 6. reparameterize (writes f32 mean / log_var into d_out, bf16 z into ws)
  {
    const int total = 8 * (int)sZL;     // 262,144
    reparam<<<dim3((total + 255) / 256), dim3(256), 0, stream>>>(
        Gb, eps, out + 1048576, out + 1310720, ZL, total);
  }

  // ---------------- decoder ----------------
  // 7. d1 = relu(A2 @ (z @ Wd0) + bd0)
  G<0,false,false>(stream, ZL, Wd0h, nullptr, T1, K2, Hh, 64, 64, Hh, Hh, sZL, 0, sK2H);
  G<2,false,false>(stream, A2, T1,   bd0,     D1, K2, Hh, K2, K2, Hh, Hh, sA2, sK2H, sK2H);
  // 8. u1 = s2 @ d1
  G<0,false,false>(stream, S2, D1, nullptr, TB, K1, Hh, K2, K2, Hh, Hh, sS2, sK2H, sK1H);
  // 9. d2 = relu(A1 @ (u1 @ Wd1) + bd1)
  G<0,false,false>(stream, TB, Wd1h, nullptr, T1, K1, Hh, Hh, Hh, Hh, Hh, sK1H, 0, sK1H);
  G<2,false,false>(stream, A1, T1,   bd1,     HP, K1, Hh, K1, K1, Hh, Hh, sA1, sK1H, sK1H);
  // 10. u2 = s1 @ d2
  G<0,false,false>(stream, S1, HP, nullptr, Z1, Nn, Hh, K1, K1, Hh, Hh, sS1, sK1H, sXH);
  // 11. d3 = relu(A0 @ (u2 @ Wd2) + bd2)
  G<0,false,false>(stream, Z1,  Wd2h, nullptr, T1, Nn, Hh, Hh, Hh, Hh, Hh, sXH, 0, sXH);
  G<2,false,false>(stream, ADJ, T1,   bd2,     H,  Nn, Hh, Nn, Nn, Hh, Hh, 0, sXH, sXH);
  // 12. out = softplus(A0 @ (d3 @ Wdf) + bdf)  -> f32 d_out
  G<0,false,false>(stream, H,   Wdfh, nullptr, T1,  Nn, 64, Hh, Hh, 64, 64, sXH, 0, sOUT);
  G<3,false,true >(stream, ADJ, T1,   bdf,     out, Nn, 64, Nn, Nn, 64, 64, 0, sOUT, sOUT);
}
